// Net_77893526880871
// MI455X (gfx1250) — matrix operand order repaired and checked by reading, not compiled, not run on hardware
//
#include <hip/hip_runtime.h>
#include <math.h>

#define NN   100000
#define NE   1600000
#define INF_ 128
#define HD   64      // heads1*hid = 8*8
#define NC   40

typedef __attribute__((ext_vector_type(16))) _Float16 v16h;
typedef __attribute__((ext_vector_type(8)))  _Float16 v8h;
typedef __attribute__((ext_vector_type(8)))  float    v8f;

// ---- monotone float <-> u32 key for atomicMax-based segment max ----
__device__ __forceinline__ unsigned fkey(float f) {
  unsigned u = __float_as_uint(f);
  return (u & 0x80000000u) ? ~u : (u | 0x80000000u);
}
__device__ __forceinline__ float funkey(unsigned k) {
  return __uint_as_float((k & 0x80000000u) ? (k & 0x7fffffffu) : ~k);
}

// =====================================================================
// GEMM1: H1[N,64] = x[N,128] @ W1[128,64]   (f16 WMMA, f32 accumulate)
// block = 256 threads (8 waves), 64 rows per block.
// =====================================================================
#define LDA1 136  // 128 + 8 halves padding (272B row -> bank-conflict-free-ish, 16B aligned)
__global__ __launch_bounds__(256) void gemm1_kernel(const float* __restrict__ x,
                                                    const float* __restrict__ W1,
                                                    float* __restrict__ H1) {
  __shared__ __align__(16) _Float16 sA[64 * LDA1];
  __shared__ __align__(16) _Float16 sB[64 * LDA1];  // W1 transposed: [n][k]
  const int tid  = threadIdx.x;
  const int brow = blockIdx.x * 64;

  // stage x tile (64x128 f32 -> f16), 2048 float4 loads
#pragma unroll
  for (int i = 0; i < 8; ++i) {
    int e4 = tid + 256 * i;
    int e  = e4 * 4;
    int r  = e >> 7, c = e & 127;
    int gr = brow + r; if (gr >= NN) gr = NN - 1;
    float4 v = *(const float4*)(x + (size_t)gr * INF_ + c);
    _Float16* p = sA + r * LDA1 + c;
    p[0] = (_Float16)v.x; p[1] = (_Float16)v.y;
    p[2] = (_Float16)v.z; p[3] = (_Float16)v.w;
  }
  // stage W1 transposed (k-major per output column)
#pragma unroll
  for (int i = 0; i < 32; ++i) {
    int e = tid + 256 * i;          // 0..8191
    int k = e >> 6, n = e & 63;
    sB[n * LDA1 + k] = (_Float16)W1[k * HD + n];
  }
  __syncthreads();

  const int w = tid >> 5, lane = tid & 31;
  const int hi = lane >> 4, m = lane & 15;
  const int rt = w & 3;     // row tile 0..3
  const int cg = w >> 2;    // column half 0..1 (2 col-tiles each)

  v8f acc0 = {}, acc1 = {};
#pragma unroll
  for (int k0 = 0; k0 < 128; k0 += 32) {
    v16h a;
    {
      const _Float16* ap = sA + (16 * rt + m) * LDA1 + k0 + 8 * hi;
      const v8h a0 = *(const v8h*)(ap);
      const v8h a1 = *(const v8h*)(ap + 16);
#pragma unroll
      for (int j = 0; j < 8; ++j) { a[j] = a0[j]; a[j + 8] = a1[j]; }
    }
#pragma unroll
    for (int ct = 0; ct < 2; ++ct) {
      const int n = 32 * cg + 16 * ct + m;
      const _Float16* bp = sB + n * LDA1 + k0 + 8 * hi;
      const v8h b0 = *(const v8h*)(bp);
      const v8h b1 = *(const v8h*)(bp + 16);
      v16h b;
#pragma unroll
      for (int j = 0; j < 8; ++j) { b[j] = b0[j]; b[j + 8] = b1[j]; }
      if (ct == 0)
        acc0 = __builtin_amdgcn_wmma_f32_16x16x32_f16(false, a, false, b, (short)0, acc0, false, false);
      else
        acc1 = __builtin_amdgcn_wmma_f32_16x16x32_f16(false, a, false, b, (short)0, acc1, false, false);
    }
  }
#pragma unroll
  for (int ct = 0; ct < 2; ++ct) {
    const v8f acc = ct ? acc1 : acc0;
    const int colg = 32 * cg + 16 * ct + m;
#pragma unroll
    for (int v = 0; v < 8; ++v) {
      int rowg = brow + 16 * rt + 8 * hi + v;
      if (rowg < NN) H1[(size_t)rowg * HD + colg] = acc[v];
    }
  }
}

// =====================================================================
// GEMM2: O[N,40] = Hact[N,64] @ W2[64,40]  (pad cols to 48)
// block = 128 threads (4 waves), 64 rows per block.
// =====================================================================
#define LDA2 72  // 64 + 8 halves padding
__global__ __launch_bounds__(128) void gemm2_kernel(const float* __restrict__ Hact,
                                                    const float* __restrict__ W2,
                                                    float* __restrict__ O) {
  __shared__ __align__(16) _Float16 sA[64 * LDA2];
  __shared__ __align__(16) _Float16 sB[48 * LDA2];  // W2 transposed, zero-padded cols
  const int tid  = threadIdx.x;
  const int brow = blockIdx.x * 64;

#pragma unroll
  for (int i = 0; i < 8; ++i) {
    int e4 = tid + 128 * i;
    int e  = e4 * 4;
    int r  = e >> 6, c = e & 63;
    int gr = brow + r; if (gr >= NN) gr = NN - 1;
    float4 v = *(const float4*)(Hact + (size_t)gr * HD + c);
    _Float16* p = sA + r * LDA2 + c;
    p[0] = (_Float16)v.x; p[1] = (_Float16)v.y;
    p[2] = (_Float16)v.z; p[3] = (_Float16)v.w;
  }
#pragma unroll
  for (int i = 0; i < 24; ++i) {
    int e = tid + 128 * i;          // 0..3071
    int n = e >> 6, k = e & 63;     // n 0..47
    float val = (n < NC) ? W2[k * NC + n] : 0.f;
    sB[n * LDA2 + k] = (_Float16)val;
  }
  __syncthreads();

  const int w = tid >> 5, lane = tid & 31;
  const int hi = lane >> 4, m = lane & 15;

#pragma unroll
  for (int ct = 0; ct < 3; ++ct) {
    v8f acc = {};
#pragma unroll
    for (int k0 = 0; k0 < 64; k0 += 32) {
      v16h a;
      {
        const _Float16* ap = sA + (16 * w + m) * LDA2 + k0 + 8 * hi;
        const v8h a0 = *(const v8h*)(ap);
        const v8h a1 = *(const v8h*)(ap + 16);
#pragma unroll
        for (int j = 0; j < 8; ++j) { a[j] = a0[j]; a[j + 8] = a1[j]; }
      }
      const int n = 16 * ct + m;
      const _Float16* bp = sB + n * LDA2 + k0 + 8 * hi;
      const v8h b0 = *(const v8h*)(bp);
      const v8h b1 = *(const v8h*)(bp + 16);
      v16h b;
#pragma unroll
      for (int j = 0; j < 8; ++j) { b[j] = b0[j]; b[j + 8] = b1[j]; }
      acc = __builtin_amdgcn_wmma_f32_16x16x32_f16(false, a, false, b, (short)0, acc, false, false);
    }
    const int colg = 16 * ct + m;
    if (colg < NC) {
#pragma unroll
      for (int v = 0; v < 8; ++v) {
        int rowg = brow + 16 * w + 8 * hi + v;
        if (rowg < NN) O[(size_t)rowg * NC + colg] = acc[v];
      }
    }
  }
}

// =====================================================================
// Per-node attention scores, layer 1: s_src[n,h], s_dst[n,h]
// =====================================================================
__global__ void nodescore1_kernel(const float* __restrict__ H1,
                                  const float* __restrict__ as,
                                  const float* __restrict__ ad,
                                  float* __restrict__ ssrc, float* __restrict__ sdst) {
  int t = blockIdx.x * blockDim.x + threadIdx.x;
  if (t >= NN * 8) return;
  int n = t >> 3, h = t & 7;
  const float* hp = H1 + (size_t)n * HD + h * 8;
  const float* a1 = as + h * 8;
  const float* a2 = ad + h * 8;
  float s1 = 0.f, s2 = 0.f;
#pragma unroll
  for (int f = 0; f < 8; ++f) { s1 += hp[f] * a1[f]; s2 += hp[f] * a2[f]; }
  ssrc[t] = s1; sdst[t] = s2;
}

// =====================================================================
// Edge passes, layer 1 (8 heads)
// =====================================================================
__global__ void edge1_max_kernel(const int* __restrict__ ei, const float* __restrict__ ea,
                                 const float* __restrict__ ssrc, const float* __restrict__ sdst,
                                 const float* __restrict__ ae,
                                 float* __restrict__ logits, unsigned* __restrict__ nmax) {
  int e = blockIdx.x * blockDim.x + threadIdx.x;
  if (e >= NE) return;
  int s = ei[e], d = ei[NE + e];
  float wgt = ea[e];
#pragma unroll
  for (int h = 0; h < 8; ++h) {
    float lg = ssrc[s * 8 + h] + sdst[d * 8 + h] + wgt * ae[h];
    lg = lg > 0.f ? lg : 0.2f * lg;                 // leaky_relu(0.2)
    logits[(size_t)e * 8 + h] = lg;
    atomicMax(nmax + d * 8 + h, fkey(lg));
  }
}

__global__ void edge1_sum_kernel(const int* __restrict__ ei,
                                 const unsigned* __restrict__ nmax,
                                 float* __restrict__ num, float* __restrict__ nsum) {
  int t = blockIdx.x * blockDim.x + threadIdx.x;
  if (t >= NE * 8) return;
  int e = t >> 3, h = t & 7;
  int d = ei[NE + e];
  float v = expf(num[t] - funkey(nmax[d * 8 + h]));
  num[t] = v;
  atomicAdd(nsum + d * 8 + h, v);
}

__global__ void edge1_scatter_kernel(const int* __restrict__ ei,
                                     const float* __restrict__ num,
                                     const float* __restrict__ nsum,
                                     const float* __restrict__ H1,
                                     float* __restrict__ Hagg) {
  int t = blockIdx.x * blockDim.x + threadIdx.x;
  if (t >= NE * 8) return;
  int e = t >> 3, h = t & 7;
  int s = ei[e], d = ei[NE + e];
  float alpha = num[t] / (nsum[d * 8 + h] + 1e-16f);
  const float* hs = H1 + (size_t)s * HD + h * 8;
  float* od = Hagg + (size_t)d * HD + h * 8;
#pragma unroll
  for (int f = 0; f < 8; ++f) atomicAdd(od + f, alpha * hs[f]);
}

// ELU in place
__global__ void elu_kernel(float* __restrict__ a, int n) {
  int t = blockIdx.x * blockDim.x + threadIdx.x;
  if (t >= n) return;
  float v = a[t];
  a[t] = v > 0.f ? v : (expf(v) - 1.f);
}

// =====================================================================
// Per-node scores, layer 2 (1 head, 40 features)
// =====================================================================
__global__ void nodescore2_kernel(const float* __restrict__ O,
                                  const float* __restrict__ as,
                                  const float* __restrict__ ad,
                                  float* __restrict__ ssrc, float* __restrict__ sdst) {
  int n = blockIdx.x * blockDim.x + threadIdx.x;
  if (n >= NN) return;
  const float* op = O + (size_t)n * NC;
  float s1 = 0.f, s2 = 0.f;
#pragma unroll
  for (int f = 0; f < NC; ++f) { float v = op[f]; s1 += v * as[f]; s2 += v * ad[f]; }
  ssrc[n] = s1; sdst[n] = s2;
}

__global__ void edge2_max_kernel(const int* __restrict__ ei, const float* __restrict__ ea,
                                 const float* __restrict__ ssrc, const float* __restrict__ sdst,
                                 const float* __restrict__ ae2,
                                 float* __restrict__ logit, unsigned* __restrict__ nmax) {
  int e = blockIdx.x * blockDim.x + threadIdx.x;
  if (e >= NE) return;
  int s = ei[e], d = ei[NE + e];
  float lg = ssrc[s] + sdst[d] + ea[e] * ae2[0];
  lg = lg > 0.f ? lg : 0.2f * lg;
  logit[e] = lg;
  atomicMax(nmax + d, fkey(lg));
}

__global__ void edge2_sum_kernel(const int* __restrict__ ei,
                                 const unsigned* __restrict__ nmax,
                                 float* __restrict__ num, float* __restrict__ nsum) {
  int e = blockIdx.x * blockDim.x + threadIdx.x;
  if (e >= NE) return;
  int d = ei[NE + e];
  float v = expf(num[e] - funkey(nmax[d]));
  num[e] = v;
  atomicAdd(nsum + d, v);
}

__global__ void edge2_scatter_kernel(const int* __restrict__ ei,
                                     const float* __restrict__ num,
                                     const float* __restrict__ nsum,
                                     const float* __restrict__ O,
                                     float* __restrict__ out) {
  long long t = (long long)blockIdx.x * blockDim.x + threadIdx.x;
  if (t >= (long long)NE * NC) return;
  int e = (int)(t / NC), f = (int)(t - (long long)e * NC);
  int s = ei[e], d = ei[NE + e];
  float alpha = num[e] / (nsum[d] + 1e-16f);
  atomicAdd(out + (size_t)d * NC + f, alpha * O[(size_t)s * NC + f]);
}

// =====================================================================
// Row-wise log-softmax in place; one wave32 per node (lane covers f, f+32)
// =====================================================================
__global__ __launch_bounds__(128) void lsm_kernel(float* __restrict__ out) {
  int node = blockIdx.x * 4 + (threadIdx.x >> 5);
  int lane = threadIdx.x & 31;
  if (node >= NN) return;
  float* row = out + (size_t)node * NC;
  float a = row[lane];
  float b = (lane < NC - 32) ? row[32 + lane] : -3.4e38f;
  float m = fmaxf(a, b);
#pragma unroll
  for (int o = 16; o > 0; o >>= 1) m = fmaxf(m, __shfl_xor(m, o, 32));
  float s = expf(a - m) + ((lane < NC - 32) ? expf(b - m) : 0.f);
#pragma unroll
  for (int o = 16; o > 0; o >>= 1) s += __shfl_xor(s, o, 32);
  float lse = m + logf(s);
  row[lane] = a - lse;
  if (lane < NC - 32) row[32 + lane] = b - lse;
}

// =====================================================================
extern "C" void kernel_launch(void* const* d_in, const int* in_sizes, int n_in,
                              void* d_out, int out_size, void* d_ws, size_t ws_size,
                              hipStream_t stream) {
  (void)in_sizes; (void)n_in; (void)out_size; (void)ws_size;
  const float* x    = (const float*)d_in[0];
  const int*   ei   = (const int*)  d_in[1];
  const float* ea   = (const float*)d_in[2];
  const float* W1   = (const float*)d_in[3];
  const float* as1  = (const float*)d_in[4];
  const float* ad1  = (const float*)d_in[5];
  const float* ae1  = (const float*)d_in[6];
  const float* W2   = (const float*)d_in[7];
  const float* as2  = (const float*)d_in[8];
  const float* ad2  = (const float*)d_in[9];
  const float* ae2  = (const float*)d_in[10];
  float* out = (float*)d_out;

  char* ws = (char*)d_ws;
  float*    H1    = (float*)   (ws + 0);           // N*64*4 = 25.6MB (reused as O2[N,40])
  float*    ss1   = (float*)   (ws + 25600000);    // N*8*4  (reused as ss2)
  float*    sd1   = (float*)   (ws + 28800000);    // N*8*4  (reused as sd2)
  unsigned* nmax1 = (unsigned*)(ws + 32000000);    // N*8*4  (reused as nmax2)
  float*    nsum1 = (float*)   (ws + 35200000);    // N*8*4  (reused as nsum2)
  float*    num1  = (float*)   (ws + 38400000);    // E*8*4 = 51.2MB (reused as num2)
  float*    Hagg  = (float*)   (ws + 89600000);    // N*64*4 = 25.6MB
  float*    O2    = H1;
  float*    ss2   = ss1;  float* sd2 = sd1;
  unsigned* nmax2 = nmax1; float* nsum2 = nsum1;
  float*    num2  = num1;

  // init accumulators (key 0 == minimal float key for segment-max)
  hipMemsetAsync(nmax1, 0, (size_t)NN * 8 * 4, stream);
  hipMemsetAsync(nsum1, 0, (size_t)NN * 8 * 4, stream);
  hipMemsetAsync(Hagg,  0, (size_t)NN * HD * 4, stream);
  hipMemsetAsync(out,   0, (size_t)NN * NC * 4, stream);

  // ---- layer 1 ----
  gemm1_kernel<<<dim3((NN + 63) / 64), dim3(256), 0, stream>>>(x, W1, H1);
  nodescore1_kernel<<<dim3((NN * 8 + 255) / 256), dim3(256), 0, stream>>>(H1, as1, ad1, ss1, sd1);
  edge1_max_kernel<<<dim3((NE + 255) / 256), dim3(256), 0, stream>>>(ei, ea, ss1, sd1, ae1, num1, nmax1);
  edge1_sum_kernel<<<dim3((NE * 8 + 255) / 256), dim3(256), 0, stream>>>(ei, nmax1, num1, nsum1);
  edge1_scatter_kernel<<<dim3((NE * 8 + 255) / 256), dim3(256), 0, stream>>>(ei, num1, nsum1, H1, Hagg);
  elu_kernel<<<dim3((NN * HD + 255) / 256), dim3(256), 0, stream>>>(Hagg, NN * HD);

  // ---- layer 2 ----
  hipMemsetAsync(nmax2, 0, (size_t)NN * 4, stream);
  hipMemsetAsync(nsum2, 0, (size_t)NN * 4, stream);
  gemm2_kernel<<<dim3((NN + 63) / 64), dim3(128), 0, stream>>>(Hagg, W2, O2);
  nodescore2_kernel<<<dim3((NN + 255) / 256), dim3(256), 0, stream>>>(O2, as2, ad2, ss2, sd2);
  edge2_max_kernel<<<dim3((NE + 255) / 256), dim3(256), 0, stream>>>(ei, ea, ss2, sd2, ae2, num2, nmax2);
  edge2_sum_kernel<<<dim3((NE + 255) / 256), dim3(256), 0, stream>>>(ei, nmax2, num2, nsum2);
  edge2_scatter_kernel<<<dim3((int)(((long long)NE * NC + 255) / 256)), dim3(256), 0, stream>>>(ei, num2, nsum2, O2, out);

  // ---- log_softmax in place ----
  lsm_kernel<<<dim3(NN / 4), dim3(128), 0, stream>>>(out);
}
